// ScaledDotProductAttention_64450279244455
// MI455X (gfx1250) — matrix operand order repaired and checked by reading, not compiled, not run on hardware
//
#include <hip/hip_runtime.h>

// ---------------------------------------------------------------------------
// Fused flash-attention with time-weight + dropout for MI455X (gfx1250).
// bf16 WMMA (v_wmma_f32_16x16x32_bf16), fp32 accumulate, base-2 fp32 softmax.
// B=4, H=16, S=1024, DK=64.  wave32, 4 waves (128 thr) per block.
// Each wave owns 16 query rows; block owns 64; grid = (S/64, B*H).
// V tiles double-buffered through GLOBAL_LOAD_ASYNC_TO_LDS_B128 (ASYNCcnt):
// tile i+1 is issued after tile i is transposed to bf16, in flight while it is consumed.
// ---------------------------------------------------------------------------

typedef __bf16 bf16_t;
typedef bf16_t v16bf __attribute__((ext_vector_type(16)));
typedef bf16_t v8bf  __attribute__((ext_vector_type(8)));
typedef float  v8f   __attribute__((ext_vector_type(8)));

constexpr int Bsz   = 4;
constexpr int Hn    = 16;
constexpr int Sq    = 1024;
constexpr int DKc   = 64;
constexpr int QTILE = 64;    // q rows per block (16 per wave)
constexpr int KTILE = 32;    // k rows per iteration
constexpr int WAVES = 4;
constexpr int VSTRB = 40;    // bf16 units per row of transposed V tile (80B: 16B-aligned, bank-spread)
constexpr int PSTR  = 40;    // bf16 units per row of P tile

static __device__ __forceinline__ bf16_t f2bf(float f) { return (bf16_t)f; }

__global__ __launch_bounds__(WAVES * 32)
void attn_tw_drop_kernel(const float* __restrict__ Qg,
                         const float* __restrict__ Kg,
                         const float* __restrict__ Vg,
                         const unsigned char* __restrict__ Mg,  // attn_mask (bool)
                         const float* __restrict__ TWg,         // time_weight
                         const float* __restrict__ DUg,         // drop uniforms
                         float* __restrict__ Og) {
    __shared__ __align__(16) float  vraw[2][KTILE * DKc];       // double-buffered raw fp32 V
    __shared__ __align__(16) bf16_t vtb[DKc * VSTRB];           // transposed bf16 V: [col][k]
    __shared__ __align__(16) bf16_t ptb[WAVES][16 * PSTR];      // per-wave bf16 P: [row][k]

    const int bh    = blockIdx.y;                  // fused (b,h)
    const int qbase = blockIdx.x * QTILE;
    const int tid   = threadIdx.x;
    const int wave  = tid >> 5;
    const int lane  = tid & 31;
    const int lhalf = lane >> 4;                   // 0 / 1
    const int l16   = lane & 15;

    const size_t bhS = (size_t)bh * Sq;

    // log2(e)/sqrt(DK): fold softmax scale + base-2 conversion into Q itself.
    const float qscale = 1.44269504088896f * 0.125f;

    // ---- A fragments for Q (16x64 bf16 -> two 16x32 chunks), pre-scaled ----
    // A layout: lane(&15)=row, elems 0..15 -> K=b..b+15,
    // with b = 16*(lane>=16).
    v16bf aq[2];
    {
        const float* qrow = Qg + (bhS + qbase + wave * 16 + l16) * DKc;
        const int b = lhalf * 16;
#pragma unroll
        for (int c = 0; c < 2; ++c) {
            float4 r0 = *(const float4*)(qrow + c * 32 + b);
            float4 r1 = *(const float4*)(qrow + c * 32 + b + 4);
            float4 r2 = *(const float4*)(qrow + c * 32 + b + 8);
            float4 r3 = *(const float4*)(qrow + c * 32 + b + 12);
            aq[c][0]=f2bf(r0.x*qscale); aq[c][1]=f2bf(r0.y*qscale);
            aq[c][2]=f2bf(r0.z*qscale); aq[c][3]=f2bf(r0.w*qscale);
            aq[c][4]=f2bf(r1.x*qscale); aq[c][5]=f2bf(r1.y*qscale);
            aq[c][6]=f2bf(r1.z*qscale); aq[c][7]=f2bf(r1.w*qscale);
            aq[c][8]=f2bf(r2.x*qscale); aq[c][9]=f2bf(r2.y*qscale);
            aq[c][10]=f2bf(r2.z*qscale); aq[c][11]=f2bf(r2.w*qscale);
            aq[c][12]=f2bf(r3.x*qscale); aq[c][13]=f2bf(r3.y*qscale);
            aq[c][14]=f2bf(r3.z*qscale); aq[c][15]=f2bf(r3.w*qscale);
        }
    }

    // ---- accumulators: O is 16x64 -> 4 C-tiles of 16x16 --------------------
    v8f o[4];
#pragma unroll
    for (int j = 0; j < 4; ++j) o[j] = (v8f){};
    float mrow[8], lrow[8];
#pragma unroll
    for (int r = 0; r < 8; ++r) { mrow[r] = -3.0e38f; lrow[r] = 0.0f; }

    bf16_t* pw = &ptb[wave][0];

    const unsigned rb0 = (unsigned)(uintptr_t)(void*)&vraw[0][0];
    const unsigned rb1 = (unsigned)(uintptr_t)(void*)&vraw[1][0];

    // async-issue one 8KB V tile: 4 x b128 per thread, ASYNCcnt-tracked
    auto issue_vtile = [&](int kbt, unsigned rbase) {
        const char* vsrc = (const char*)(Vg + (bhS + kbt) * DKc);
#pragma unroll
        for (int rnd = 0; rnd < 4; ++rnd) {
            const unsigned off   = (unsigned)(tid * 16 + rnd * 2048);
            const unsigned laddr = rbase + off;
            const unsigned long long gaddr =
                (unsigned long long)(uintptr_t)(vsrc + off);
            asm volatile("global_load_async_to_lds_b128 %0, %1, off"
                         :: "v"(laddr), "v"(gaddr) : "memory");
        }
    };

    // prologue: put tile 0 in flight
    issue_vtile(0, rb0);

    for (int kb = 0; kb < Sq; kb += KTILE) {
        const int cur = (kb / KTILE) & 1;

        // ---- drain this tile's async copy, make it block-visible ----------
        asm volatile("s_wait_asynccnt 0" ::: "memory");
        __syncthreads();   // raw[cur] visible; prior vtb readers also done

        // ---- convert + transpose: vraw[cur][k][col] -> vtb[col][k] --------
        {
            const float* raw  = &vraw[cur][0];
            const int col   = tid & 63;            // one column per thread
            const int khalf = tid >> 6;            // 16 k-values per thread
            v8bf h0, h1;
#pragma unroll
            for (int e = 0; e < 8; ++e)
                h0[e] = f2bf(raw[(khalf * 16 + e) * DKc + col]);
#pragma unroll
            for (int e = 0; e < 8; ++e)
                h1[e] = f2bf(raw[(khalf * 16 + 8 + e) * DKc + col]);
            *(v8bf*)&vtb[col * VSTRB + khalf * 16]     = h0;
            *(v8bf*)&vtb[col * VSTRB + khalf * 16 + 8] = h1;
        }
        __syncthreads();   // bf16 V tile ready

        // ---- prefetch pipeline: next V tile async, next K rows hinted -----
        if (kb + KTILE < Sq) {
            issue_vtile(kb + KTILE, cur ? rb0 : rb1);
            __builtin_prefetch(Kg + (bhS + kb + KTILE + lane) * DKc, 0, 1);
        }

        // ---- scores (log2 domain): S[t] = Qs(16x64) x K_tile^T ------------
        v8f sc[2];
#pragma unroll
        for (int t = 0; t < 2; ++t) {
            v8f acc = (v8f){};
#pragma unroll
            for (int c = 0; c < 2; ++c) {
                // B layout: lane -> col N=l16, K = lhalf*16 + e
                const float* kptr = Kg + (bhS + kb + t * 16 + l16) * DKc
                                       + c * 32 + lhalf * 16;
                float4 b0 = *(const float4*)(kptr);
                float4 b1 = *(const float4*)(kptr + 4);
                float4 b2 = *(const float4*)(kptr + 8);
                float4 b3 = *(const float4*)(kptr + 12);
                v16bf bk;
                bk[0]=f2bf(b0.x); bk[1]=f2bf(b0.y); bk[2]=f2bf(b0.z); bk[3]=f2bf(b0.w);
                bk[4]=f2bf(b1.x); bk[5]=f2bf(b1.y); bk[6]=f2bf(b1.z); bk[7]=f2bf(b1.w);
                bk[8]=f2bf(b2.x); bk[9]=f2bf(b2.y); bk[10]=f2bf(b2.z); bk[11]=f2bf(b2.w);
                bk[12]=f2bf(b3.x); bk[13]=f2bf(b3.y); bk[14]=f2bf(b3.z); bk[15]=f2bf(b3.w);
                acc = __builtin_amdgcn_wmma_f32_16x16x32_bf16(
                          false, aq[c], false, bk, (short)0, acc, false, false);
            }
            sc[t] = acc;
        }

        // ---- online softmax (base-2) + mask / time_weight / dropout -------
        // ptb is wave-private; DS ops are in-order per wave -> fence with
        // s_wait_dscnt before reuse, no workgroup barrier needed.
        asm volatile("s_wait_dscnt 0" ::: "memory");
        __builtin_amdgcn_wave_barrier();

#pragma unroll
        for (int r = 0; r < 8; ++r) {
            const int qrow = qbase + wave * 16 + r + lhalf * 8;
            const size_t rb = (bhS + qrow) * (size_t)Sq + kb;
            const size_t i0 = rb + l16;
            const size_t i1 = rb + 16 + l16;

            float s0 = sc[0][r];
            float s1 = sc[1][r];
            if (Mg[i0]) s0 = -1.0e9f;              // log2-domain -inf proxy
            if (Mg[i1]) s1 = -1.0e9f;
            const float tw0 = TWg[i0], tw1 = TWg[i1];
            const float k0 = (DUg[i0] >= 0.5f) ? 2.0f : 0.0f;  // keep/(1-p)
            const float k1 = (DUg[i1] >= 0.5f) ? 2.0f : 0.0f;

            float rmax = fmaxf(s0, s1);
#pragma unroll
            for (int d = 1; d < 16; d <<= 1)
                rmax = fmaxf(rmax, __shfl_xor(rmax, d, 32));
            const float mnew = fmaxf(mrow[r], rmax);
            const float f    = exp2f(mrow[r] - mnew);   // bare v_exp_f32
            const float p0   = exp2f(s0 - mnew);
            const float p1   = exp2f(s1 - mnew);
            float ps = p0 + p1;
#pragma unroll
            for (int d = 1; d < 16; d <<= 1) ps += __shfl_xor(ps, d, 32);
            lrow[r] = lrow[r] * f + ps;
            mrow[r] = mnew;
#pragma unroll
            for (int j = 0; j < 4; ++j) o[j][r] = o[j][r] * f;

            // numerator weights -> wave-private bf16 transpose tile
            const int prow = r + lhalf * 8;
            pw[prow * PSTR + l16]      = f2bf(p0 * tw0 * k0);
            pw[prow * PSTR + 16 + l16] = f2bf(p1 * tw1 * k1);
        }

        asm volatile("s_wait_dscnt 0" ::: "memory");
        __builtin_amdgcn_wave_barrier();

        // ---- P fragment (16x32 bf16 A matrix): two 16B LDS vector loads ---
        v16bf pa;
        {
            const int b = lhalf * 16;
            const bf16_t* prow = pw + l16 * PSTR;
            v8bf lo = *(const v8bf*)(prow + b);
            v8bf hi = *(const v8bf*)(prow + b + 8);
            pa = __builtin_shufflevector(lo, hi, 0,1,2,3,4,5,6,7,
                                                 8,9,10,11,12,13,14,15);
        }

        // ---- O[j] += P(16x32) x V_tile(32x16) for each 16-col chunk -------
#pragma unroll
        for (int j = 0; j < 4; ++j) {
            const int col = j * 16 + l16;
            const bf16_t* vcol = &vtb[col * VSTRB + lhalf * 16];
            v8bf lo = *(const v8bf*)(vcol);
            v8bf hi = *(const v8bf*)(vcol + 8);
            v16bf bv = __builtin_shufflevector(lo, hi, 0,1,2,3,4,5,6,7,
                                                       8,9,10,11,12,13,14,15);
            o[j] = __builtin_amdgcn_wmma_f32_16x16x32_bf16(
                       false, pa, false, bv, (short)0, o[j], false, false);
        }
    }

    // ---- normalize (one reciprocal per row) and store ----------------------
#pragma unroll
    for (int r = 0; r < 8; ++r) {
        const float inv = 1.0f / lrow[r];
        const int qrow  = qbase + wave * 16 + r + lhalf * 8;
        float* orow = Og + (bhS + qrow) * DKc + l16;
#pragma unroll
        for (int j = 0; j < 4; ++j)
            orow[j * 16] = o[j][r] * inv;
    }
}

extern "C" void kernel_launch(void* const* d_in, const int* in_sizes, int n_in,
                              void* d_out, int out_size, void* d_ws, size_t ws_size,
                              hipStream_t stream) {
    (void)in_sizes; (void)n_in; (void)d_ws; (void)ws_size; (void)out_size;
    const float*         Q  = (const float*)d_in[0];
    const float*         K  = (const float*)d_in[1];
    const float*         V  = (const float*)d_in[2];
    const unsigned char* M  = (const unsigned char*)d_in[3];  // jnp.bool_ = 1 byte
    const float*         TW = (const float*)d_in[4];
    const float*         DU = (const float*)d_in[5];
    float*               O  = (float*)d_out;

    dim3 grid(Sq / QTILE, Bsz * Hn);
    dim3 block(WAVES * 32);
    attn_tw_drop_kernel<<<grid, block, 0, stream>>>(Q, K, V, M, TW, DU, O);
}
